// RelationConvFusion_5102421148357
// MI455X (gfx1250) — matrix operand order repaired and checked by reading, not compiled, not run on hardware
//
#include <hip/hip_runtime.h>

#define DD 256  // feature dim

typedef __attribute__((ext_vector_type(16))) __bf16        v16bf;
typedef __attribute__((ext_vector_type(8)))  float         v8f;
typedef __attribute__((ext_vector_type(4)))  unsigned int  v4u;

union Pack32 { v4u u[2]; v16bf v; };   // 32 bytes = 16 bf16

__device__ __forceinline__ unsigned short f32_bf16(float f) {
  unsigned int u = __float_as_uint(f);
  u += 0x7FFFu + ((u >> 16) & 1u);     // round-to-nearest-even
  return (unsigned short)(u >> 16);
}

// ---------------- conversion kernels ----------------
__global__ __launch_bounds__(256) void cvt_x_kernel(const float* __restrict__ x,
                                                    unsigned short* __restrict__ xb,
                                                    int total) {
  int i = blockIdx.x * 256 + threadIdx.x;
  if (i < total) xb[i] = f32_bf16(x[i]);
}

// WT[j][k] = concat(W,U,V)[k][j]  (transposed so WMMA B loads are contiguous in K)
__global__ __launch_bounds__(256) void cvt_w_kernel(const float* __restrict__ W,
                                                    const float* __restrict__ U,
                                                    const float* __restrict__ V,
                                                    unsigned short* __restrict__ WT) {
  int j = blockIdx.x;           // 0..767
  int k = threadIdx.x;          // 0..255
  const float* M = (j < 256) ? W : (j < 512) ? U : V;
  int jj = j & 255;
  WT[(size_t)j * DD + k] = f32_bf16(M[(size_t)k * DD + jj]);
}

// ---------------- fused WMMA GEMM: [Wh*al | Wh*ar | Uh | Vh] ----------------
// One wave computes a 16-row x 64-col tile: 4 x v_wmma_f32_16x16x32_bf16 per K-step,
// 8 K-steps, A fragment reused across the 4 column sub-tiles. A/B fragments are
// explicitly double-buffered so k+1's global_load_b128s are in flight while k's
// WMMAs execute (waits become loadcnt<=N instead of 0).
__global__ __launch_bounds__(256) void gemm_wmma_kernel(
    const unsigned short* __restrict__ xb,   // [N,256] bf16
    const unsigned short* __restrict__ WT,   // [768,256] bf16 (transposed weights)
    const float* __restrict__ W_b, const float* __restrict__ U_b,
    const float* __restrict__ V_b,
    const float* __restrict__ attn_l, const float* __restrict__ attn_r,
    float* __restrict__ a_src, float* __restrict__ a_dst,
    float* __restrict__ Uh, float* __restrict__ Vh,
    int rowTiles) {
  const int wavesTotal = rowTiles * 12;              // 12 col groups of 64 (768 cols)
  int wave = blockIdx.x * 8 + (threadIdx.x >> 5);
  if (wave >= wavesTotal) return;                    // wave-uniform; EXEC all-1s below
  int rowTile  = wave / 12;
  int colGroup = wave % 12;
  int lane = threadIdx.x & 31;
  int m = lane & 15;                                 // A row / B col / D col
  int h = lane >> 4;                                 // lane half

  const unsigned short* xrow  = xb + (size_t)(rowTile * 16 + m) * DD;
  const unsigned short* wbase = WT + (size_t)(colGroup * 64 + m) * DD;

  v8f acc[4] = {v8f{}, v8f{}, v8f{}, v8f{}};

  // A 16x32 bf16 fragment: elems 0..7 = K(8h..8h+7), elems 8..15 = K(16+8h..+7)
  // B 32x16 bf16 fragment: same K order as A: elems 0..7 = K(8h..8h+7), elems 8..15 = K(16+8h..+7)
  Pack32 aCur, aNxt;
  Pack32 bCur[4], bNxt[4];

  const unsigned short* ap = xrow + 8 * h;
  const unsigned short* bp = wbase + 8 * h;

  aCur.u[0] = *(const v4u*)(ap);
  aCur.u[1] = *(const v4u*)(ap + 16);
#pragma unroll
  for (int s = 0; s < 4; ++s) {
    const unsigned short* w = bp + (size_t)s * 16 * DD;
    bCur[s].u[0] = *(const v4u*)(w);
    bCur[s].u[1] = *(const v4u*)(w + 16);
  }

#pragma unroll
  for (int kt = 0; kt < 8; ++kt) {
    if (kt < 7) {                                    // prefetch k+1 fragments
      int k1 = (kt + 1) * 32;
      aNxt.u[0] = *(const v4u*)(ap + k1);
      aNxt.u[1] = *(const v4u*)(ap + k1 + 16);
#pragma unroll
      for (int s = 0; s < 4; ++s) {
        const unsigned short* w = bp + (size_t)s * 16 * DD + k1;
        bNxt[s].u[0] = *(const v4u*)(w);
        bNxt[s].u[1] = *(const v4u*)(w + 16);
      }
    }
#pragma unroll
    for (int s = 0; s < 4; ++s)
      acc[s] = __builtin_amdgcn_wmma_f32_16x16x32_bf16(
          false, aCur.v, false, bCur[s].v, (short)0, acc[s], false, false);
    if (kt < 7) {
      aCur = aNxt;
#pragma unroll
      for (int s = 0; s < 4; ++s) bCur[s] = bNxt[s];
    }
  }

  // Epilogue: D element acc[s][r] -> row = rowTile*16 + 8h + r, col = colGroup*64+s*16+m
  int baseRow = rowTile * 16 + 8 * h;
#pragma unroll
  for (int s = 0; s < 4; ++s) {
    int j = colGroup * 64 + s * 16 + m;
    if (j < 256) {                               // Wh block -> pre-scale by attn vectors
      float bias = W_b[j], al = attn_l[j], ar = attn_r[j];
#pragma unroll
      for (int r = 0; r < 8; ++r) {
        float v = acc[s][r] + bias;
        size_t o = (size_t)(baseRow + r) * DD + j;
        a_src[o] = v * al;
        a_dst[o] = v * ar;
      }
    } else if (j < 512) {                        // Uh block
      int jj = j - 256;
      float bias = U_b[jj];
#pragma unroll
      for (int r = 0; r < 8; ++r)
        Uh[(size_t)(baseRow + r) * DD + jj] = acc[s][r] + bias;
    } else {                                     // Vh block
      int jj = j - 512;
      float bias = V_b[jj];
#pragma unroll
      for (int r = 0; r < 8; ++r)
        Vh[(size_t)(baseRow + r) * DD + jj] = acc[s][r] + bias;
    }
  }
}

// ---------------- per-edge gate + scatter-add (one wave32 per edge) ----------------
// Hardware no-return FP32 atomic (STOREcnt-tracked); guarantees global_atomic_add_f32
// instead of a possible CAS-loop expansion of the generic atomic builtin.
__device__ __forceinline__ void atom_add(float* p, float v) {
  asm volatile("global_atomic_add_f32 %0, %1, off"
               :
               : "v"(p), "v"(v)
               : "memory");
}

__global__ __launch_bounds__(256) void edge_gate_kernel(
    const int* __restrict__ src, const int* __restrict__ dst,
    const float* __restrict__ a_src, const float* __restrict__ a_dst,
    const float* __restrict__ Vh,
    float* __restrict__ ssh, float* __restrict__ ssum, int E) {
  int e = blockIdx.x * 8 + (threadIdx.x >> 5);
  if (e >= E) return;
  int lane = threadIdx.x & 31;
  int s = src[e], t = dst[e];
  const float4* ps = (const float4*)(a_src + (size_t)s * DD);
  const float4* pt = (const float4*)(a_dst + (size_t)t * DD);
  const float4* pv = (const float4*)(Vh    + (size_t)s * DD);
  float* osh = ssh  + (size_t)t * DD;
  float* oss = ssum + (size_t)t * DD;
#pragma unroll
  for (int i = 0; i < 2; ++i) {
    int q = lane + 32 * i;                 // float4 index, 64 per row
    float4 as = ps[q], ad = pt[q], vv = pv[q];
    float gx = 1.f / (1.f + __expf(-(as.x + ad.x)));
    float gy = 1.f / (1.f + __expf(-(as.y + ad.y)));
    float gz = 1.f / (1.f + __expf(-(as.z + ad.z)));
    float gw = 1.f / (1.f + __expf(-(as.w + ad.w)));
    int b = 4 * q;
    atom_add(oss + b + 0, gx);  atom_add(osh + b + 0, gx * vv.x);
    atom_add(oss + b + 1, gy);  atom_add(osh + b + 1, gy * vv.y);
    atom_add(oss + b + 2, gz);  atom_add(osh + b + 2, gz * vv.z);
    atom_add(oss + b + 3, gw);  atom_add(osh + b + 3, gw * vv.w);
  }
}

// ---------------- per-node combine + LayerNorm + ReLU + residual ----------------
__global__ __launch_bounds__(256) void finalize_kernel(
    const float* __restrict__ x, const float* __restrict__ Uh,
    const float* __restrict__ ssh, const float* __restrict__ ssum,
    const float* __restrict__ gamma, const float* __restrict__ beta,
    float* __restrict__ out) {
  int row = blockIdx.x, d = threadIdx.x;
  size_t o = (size_t)row * DD + d;
  float hv = Uh[o] + ssh[o] / (ssum[o] + 1e-6f);

  float s1 = hv, s2 = hv * hv;
#pragma unroll
  for (int off = 16; off; off >>= 1) {     // wave32 reduction
    s1 += __shfl_xor(s1, off, 32);
    s2 += __shfl_xor(s2, off, 32);
  }
  __shared__ float r1[8], r2[8];
  int wid = threadIdx.x >> 5;
  if ((threadIdx.x & 31) == 0) { r1[wid] = s1; r2[wid] = s2; }
  __syncthreads();
  float t1 = 0.f, t2 = 0.f;
#pragma unroll
  for (int i = 0; i < 8; ++i) { t1 += r1[i]; t2 += r2[i]; }

  float mu  = t1 * (1.0f / DD);
  float var = t2 * (1.0f / DD) - mu * mu;
  float nrm = (hv - mu) * rsqrtf(var + 1e-6f) * gamma[d] + beta[d];
  out[o] = x[o] + fmaxf(nrm, 0.f);
}

// ---------------- host side ----------------
extern "C" void kernel_launch(void* const* d_in, const int* in_sizes, int n_in,
                              void* d_out, int out_size, void* d_ws, size_t ws_size,
                              hipStream_t stream) {
  const float* x      = (const float*)d_in[0];
  const int*   src    = (const int*)d_in[1];   // JAX default x64-off -> int32
  const int*   dst    = (const int*)d_in[2];
  const float* W_w    = (const float*)d_in[3];
  const float* W_b    = (const float*)d_in[4];
  const float* U_w    = (const float*)d_in[5];
  const float* U_b    = (const float*)d_in[6];
  const float* V_w    = (const float*)d_in[7];
  const float* V_b    = (const float*)d_in[8];
  const float* attn_l = (const float*)d_in[9];
  const float* attn_r = (const float*)d_in[10];
  const float* gamma  = (const float*)d_in[11];
  const float* beta   = (const float*)d_in[12];

  int N = in_sizes[0] / DD;
  int E = in_sizes[1];
  size_t ND = (size_t)N * DD;

  char* ws = (char*)d_ws;
  size_t off = 0;
  auto alloc = [&](size_t bytes) -> void* {
    void* p = ws + off;
    off = (off + bytes + 255) & ~(size_t)255;
    return p;
  };
  unsigned short* xb    = (unsigned short*)alloc(ND * 2);
  unsigned short* WT    = (unsigned short*)alloc((size_t)768 * DD * 2);
  float*          a_src = (float*)alloc(ND * 4);
  float*          a_dst = (float*)alloc(ND * 4);
  float*          Uh    = (float*)alloc(ND * 4);
  float*          Vh    = (float*)alloc(ND * 4);
  float*          ssh   = (float*)alloc(ND * 4);
  float*          ssum  = (float*)alloc(ND * 4);

  hipMemsetAsync(ssh,  0, ND * 4, stream);
  hipMemsetAsync(ssum, 0, ND * 4, stream);

  int totX = (int)ND;
  cvt_x_kernel<<<(totX + 255) / 256, 256, 0, stream>>>(x, xb, totX);
  cvt_w_kernel<<<768, 256, 0, stream>>>(W_w, U_w, V_w, WT);

  int rowTiles = (N + 15) / 16;                 // N=50000 -> exactly 3125
  int waves = rowTiles * 12;
  gemm_wmma_kernel<<<(waves + 7) / 8, 256, 0, stream>>>(
      xb, WT, W_b, U_b, V_b, attn_l, attn_r, a_src, a_dst, Uh, Vh, rowTiles);

  edge_gate_kernel<<<(E + 7) / 8, 256, 0, stream>>>(
      src, dst, a_src, a_dst, Vh, ssh, ssum, E);

  finalize_kernel<<<N, 256, 0, stream>>>(x, Uh, ssh, ssum, gamma, beta,
                                         (float*)d_out);
}
